// RingAttention_43568148251093
// MI455X (gfx1250) — compile-verified, matrix operand order checked by reading, not run on hardware
//
#include <hip/hip_runtime.h>

typedef __attribute__((ext_vector_type(16))) __bf16 v16bf;
typedef __attribute__((ext_vector_type(8)))  __bf16 v8bf;
typedef __attribute__((ext_vector_type(8)))  float  v8f;

namespace {

constexpr int kB   = 2;
constexpr int kS   = 2048;
constexpr int kNH  = 16;
constexpr int kHD  = 128;
constexpr int kTK  = 32;                 // keys per tile
constexpr int kWaves = 8;
constexpr int kQRowsBlock = 16 * kWaves; // 128 query rows per block
constexpr int kRowStride = kNH * kHD;    // 2048 floats per (b,s) row
constexpr int kNT  = kS / kTK;           // 64 key tiles

__device__ inline v16bf load_frag16(const __bf16* lo, const __bf16* hi) {
  v8bf a = *reinterpret_cast<const v8bf*>(lo);
  v8bf b = *reinterpret_cast<const v8bf*>(hi);
  v16bf r;
#pragma unroll
  for (int i = 0; i < 8; ++i) { r[i] = a[i]; r[i + 8] = b[i]; }
  return r;
}

struct Stage {
  float4 k[4];
  float4 v[4];
};

__device__ inline void stage_load(Stage& st, const float* kT, const float* vT,
                                  int tid) {
#pragma unroll
  for (int it = 0; it < 4; ++it) {
    const int u   = tid + it * 256;   // vec4 unit id, 1024 total
    const int key = u >> 5;
    const int col = (u & 31) * 4;
    st.k[it] = *(const float4*)(kT + (size_t)key * kRowStride + col);
    st.v[it] = *(const float4*)(vT + (size_t)key * kRowStride + col);
  }
}

__device__ inline void stage_store(const Stage& st, __bf16* dK, __bf16* dVt,
                                   int tid) {
#pragma unroll
  for (int it = 0; it < 4; ++it) {
    const int u   = tid + it * 256;
    const int key = u >> 5;
    const int col = (u & 31) * 4;
    __bf16* kd = &dK[key * kHD + col];
    kd[0] = (__bf16)st.k[it].x; kd[1] = (__bf16)st.k[it].y;
    kd[2] = (__bf16)st.k[it].z; kd[3] = (__bf16)st.k[it].w;
    dVt[(col + 0) * kTK + key] = (__bf16)st.v[it].x;
    dVt[(col + 1) * kTK + key] = (__bf16)st.v[it].y;
    dVt[(col + 2) * kTK + key] = (__bf16)st.v[it].z;
    dVt[(col + 3) * kTK + key] = (__bf16)st.v[it].w;
  }
}

} // namespace

__global__ __launch_bounds__(256, 1) void ring_attn_fused(
    const float* __restrict__ Qg, const float* __restrict__ Kg,
    const float* __restrict__ Vg, float* __restrict__ Og) {
  __shared__ __align__(16) __bf16 sK[2][kTK * kHD];     // (key, d) row-major
  __shared__ __align__(16) __bf16 sVt[2][kHD * kTK];    // (d, key) transposed
  __shared__ __align__(16) __bf16 sP[kWaves][16 * kTK]; // per-wave P bounce

  const int tid  = threadIdx.x;
  const int wave = tid >> 5;
  const int lane = tid & 31;
  const int half = lane >> 4;
  const int l16  = lane & 15;

  const int bid  = blockIdx.x;
  const int qblk = bid & 15;
  const int h    = (bid >> 4) & 15;
  const int b    = bid >> 8;

  const size_t headOff = ((size_t)b * kS * kNH + h) * (size_t)kHD;
  const float* qb = Qg + headOff;
  const float* kb = Kg + headOff;
  const float* vb = Vg + headOff;
  float*       ob = Og + headOff;

  const int qrow0 = qblk * kQRowsBlock + wave * 16;

  // ---- Q tile -> 4 bf16 A-fragments, pre-scaled by (1/sqrt(hd)) * log2(e)
  const float kScale = 0.08838834764831845f * 1.4426950408889634f;
  v16bf qf[4];
  {
    const float* qr = qb + (size_t)(qrow0 + l16) * kRowStride;
#pragma unroll
    for (int t = 0; t < 4; ++t) {
      const int k0 = t * 32 + half * 8;
      float tmp[16];
      *(float4*)&tmp[0]  = *(const float4*)(qr + k0);
      *(float4*)&tmp[4]  = *(const float4*)(qr + k0 + 4);
      *(float4*)&tmp[8]  = *(const float4*)(qr + k0 + 16);
      *(float4*)&tmp[12] = *(const float4*)(qr + k0 + 20);
#pragma unroll
      for (int i = 0; i < 16; ++i) qf[t][i] = (__bf16)(tmp[i] * kScale);
    }
  }

  v16bf vone;   // all-ones B fragment: rowsum via WMMA instead of 32 shuffles
#pragma unroll
  for (int i = 0; i < 16; ++i) vone[i] = (__bf16)1.0f;

  v8f oacc[8];
  float mrun[8], lrun[8];
#pragma unroll
  for (int c = 0; c < 8; ++c)
#pragma unroll
    for (int r = 0; r < 8; ++r) oacc[c][r] = 0.0f;
#pragma unroll
  for (int r = 0; r < 8; ++r) { mrun[r] = -1.0e30f; lrun[r] = 0.0f; }

  // ---- prologue: stage tile 0 into buffer 0
  Stage st;
  stage_load(st, kb, vb, tid);
  stage_store(st, sK[0], sVt[0], tid);
  __syncthreads();

  int buf = 0;
  for (int kt = 0; kt < kNT; ++kt) {
    const bool more = (kt + 1 < kNT);
    if (more) {  // issue next tile's global loads early; in flight during WMMAs
      stage_load(st, kb + (size_t)(kt + 1) * kTK * kRowStride,
                 vb + (size_t)(kt + 1) * kTK * kRowStride, tid);
    }

    // ---- scores: S = Qs * K^T, fragments hoisted in groups of 4
    v8f s0, s1;
#pragma unroll
    for (int r = 0; r < 8; ++r) { s0[r] = 0.0f; s1[r] = 0.0f; }
    {
      v16bf kf[4];
#pragma unroll
      for (int t = 0; t < 4; ++t) {
        const __bf16* p = &sK[buf][(0 * 16 + l16) * kHD + t * 32 + half * 8];
        kf[t] = load_frag16(p, p + 16);
      }
#pragma unroll
      for (int t = 0; t < 4; ++t)
        s0 = __builtin_amdgcn_wmma_f32_16x16x32_bf16(false, qf[t], false, kf[t],
                                                     (short)0, s0, false, false);
#pragma unroll
      for (int t = 0; t < 4; ++t) {
        const __bf16* p = &sK[buf][(1 * 16 + l16) * kHD + t * 32 + half * 8];
        kf[t] = load_frag16(p, p + 16);
      }
#pragma unroll
      for (int t = 0; t < 4; ++t)
        s1 = __builtin_amdgcn_wmma_f32_16x16x32_bf16(false, qf[t], false, kf[t],
                                                     (short)0, s1, false, false);
    }

    // ---- convert + store next tile while scores settle (fills hazard slots)
    if (more) stage_store(st, sK[buf ^ 1], sVt[buf ^ 1], tid);

    // ---- online softmax (exp2 domain; row r+8*half lives in register r)
    float mx[8];
#pragma unroll
    for (int r = 0; r < 8; ++r) mx[r] = fmaxf(s0[r], s1[r]);
#pragma unroll
    for (int m = 1; m <= 8; m <<= 1)
#pragma unroll
      for (int r = 0; r < 8; ++r) mx[r] = fmaxf(mx[r], __shfl_xor(mx[r], m, 32));

    float corr[8], p0[8], p1[8];
#pragma unroll
    for (int r = 0; r < 8; ++r) {
      const float mn = fmaxf(mrun[r], mx[r]);
      corr[r] = __builtin_amdgcn_exp2f(mrun[r] - mn);
      mrun[r] = mn;
      p0[r] = __builtin_amdgcn_exp2f(s0[r] - mn);
      p1[r] = __builtin_amdgcn_exp2f(s1[r] - mn);
    }
#pragma unroll
    for (int c = 0; c < 8; ++c)
#pragma unroll
      for (int r = 0; r < 8; ++r) oacc[c][r] *= corr[r];

    // ---- P: C-layout -> LDS bounce -> A-fragment (wave-private region)
    __bf16* pw = &sP[wave][0];
#pragma unroll
    for (int r = 0; r < 8; ++r) {
      const int row = r + 8 * half;
      pw[row * kTK + l16]      = (__bf16)p0[r];
      pw[row * kTK + 16 + l16] = (__bf16)p1[r];
    }
    asm volatile("s_wait_dscnt 0" ::: "memory");  // in-order LDS pipe
    const __bf16* pr = &sP[wave][l16 * kTK + half * 8];
    v16bf pa = load_frag16(pr, pr + 16);

    // ---- rowsum via WMMA (P x ones): lands per-register like lrun
    v8f zc;
#pragma unroll
    for (int r = 0; r < 8; ++r) zc[r] = 0.0f;
    v8f rsum = __builtin_amdgcn_wmma_f32_16x16x32_bf16(false, pa, false, vone,
                                                       (short)0, zc, false,
                                                       false);

    // ---- O += P * V, fragments hoisted in groups of 4
    {
      v16bf vf[4];
#pragma unroll
      for (int c = 0; c < 4; ++c) {
        const __bf16* p = &sVt[buf][(c * 16 + l16) * kTK + half * 8];
        vf[c] = load_frag16(p, p + 16);
      }
#pragma unroll
      for (int c = 0; c < 4; ++c)
        oacc[c] = __builtin_amdgcn_wmma_f32_16x16x32_bf16(false, pa, false,
                                                          vf[c], (short)0,
                                                          oacc[c], false, false);
#pragma unroll
      for (int c = 0; c < 4; ++c) {
        const __bf16* p = &sVt[buf][((c + 4) * 16 + l16) * kTK + half * 8];
        vf[c] = load_frag16(p, p + 16);
      }
#pragma unroll
      for (int c = 0; c < 4; ++c)
        oacc[c + 4] = __builtin_amdgcn_wmma_f32_16x16x32_bf16(false, pa, false,
                                                              vf[c], (short)0,
                                                              oacc[c + 4],
                                                              false, false);
    }

#pragma unroll
    for (int r = 0; r < 8; ++r) lrun[r] = lrun[r] * corr[r] + rsum[r];

    __syncthreads();  // staging of buf^1 done; compute on buf done
    buf ^= 1;
  }

  // ---- normalize + store (B, S, nh, hd) layout, fp32
  float inv[8];
#pragma unroll
  for (int r = 0; r < 8; ++r) inv[r] = 1.0f / lrun[r];
#pragma unroll
  for (int c = 0; c < 8; ++c) {
#pragma unroll
    for (int r = 0; r < 8; ++r) {
      const int qrow = qrow0 + r + 8 * half;
      ob[(size_t)qrow * kRowStride + c * 16 + l16] = oacc[c][r] * inv[r];
    }
  }
}

extern "C" void kernel_launch(void* const* d_in, const int* in_sizes, int n_in,
                              void* d_out, int out_size, void* d_ws, size_t ws_size,
                              hipStream_t stream) {
  const float* Q = (const float*)d_in[0];
  const float* K = (const float*)d_in[1];
  const float* V = (const float*)d_in[2];
  float* O = (float*)d_out;
  const int grid = kB * kNH * (kS / kQRowsBlock);  // 2*16*16 = 512 blocks
  ring_attn_fused<<<grid, 256, 0, stream>>>(Q, K, V, O);
}
